// MultiHeadAttention_29532195127778
// MI455X (gfx1250) — matrix operand order repaired and checked by reading, not compiled, not run on hardware
//
#include <hip/hip_runtime.h>
#include <hip/hip_bf16.h>
#include <math.h>

// MI455X / gfx1250: wave32, WMMA bf16 16x16x32, 320KB LDS per WGP, TDM async copy.
typedef __attribute__((ext_vector_type(16))) __bf16 v16bf;
typedef __attribute__((ext_vector_type(8)))  __bf16 v8bf;
typedef __attribute__((ext_vector_type(4)))  __bf16 v4bf;
typedef __attribute__((ext_vector_type(8)))  float  v8f;
typedef __attribute__((ext_vector_type(4)))  float  v4f;
typedef __attribute__((ext_vector_type(4)))  unsigned int u32x4;
typedef __attribute__((ext_vector_type(8)))  int    i32x8;
typedef __attribute__((ext_vector_type(4)))  int    i32x4;

#define B_   4
#define S_   2048
#define DM_  1024
#define H_   8
#define QT   16
#define NEG_INF_SCALE (-1e9f)

// LDS layout (dynamic): 16x1024 bf16 Q | 16x2048 f32 scores | 16x2048 bf16 W | reductions
#define SQ_BYTES   (QT * DM_ * 2)              // 32768
#define SS_BYTES   (QT * S_  * 4)              // 131072
#define SW_BYTES   (QT * S_  * 2)              // 65536
#define RED_BYTES  ((256 + 32) * 4)            // 1152
#define SMEM_BYTES (SQ_BYTES + SS_BYTES + SW_BYTES + RED_BYTES)

#if defined(__has_builtin)
#  if __has_builtin(__builtin_amdgcn_tensor_load_to_lds) && \
      __has_builtin(__builtin_amdgcn_s_wait_tensorcnt)
#    define HAVE_TDM 1
#  endif
#endif
#ifndef HAVE_TDM
#  define HAVE_TDM 0
#endif

__device__ __forceinline__ __bf16 cvt_bf16(float x)  { return (__bf16)x; }
__device__ __forceinline__ __bf16 cvt_bf16(__bf16 x) { return x; }

// A-fragment (16x32 bf16) from LDS: lane&15 = row; lanes>=16 shift K by +16.
// halves 0..7 -> K base+0..7, halves 8..15 -> K base+8..15 (two b128 LDS loads).
__device__ __forceinline__ v16bf frag_a_lds(const __bf16* p) {
    v8bf lo = *(const v8bf*)(p);
    v8bf hi = *(const v8bf*)(p + 8);
    v16bf r;
#pragma unroll
    for (int i = 0; i < 8; ++i) { r[i] = lo[i]; r[i + 8] = hi[i]; }
    return r;
}

// B-fragment (32x16 bf16): lane&15 = column; lane-group picks K 0..15 vs 16..31,
// contiguous in one operand row. f32 source: 4x b128 + cvt; bf16 source: 2x b128.
__device__ __forceinline__ v16bf frag_b_contig(const float* p) {
    v4f x0 = *(const v4f*)(p);
    v4f x1 = *(const v4f*)(p + 4);
    v4f x2 = *(const v4f*)(p + 8);
    v4f x3 = *(const v4f*)(p + 12);
    v16bf r;
#pragma unroll
    for (int i = 0; i < 4; ++i) {
        r[i]      = (__bf16)x0[i];
        r[i + 4]  = (__bf16)x1[i];
        r[i + 8]  = (__bf16)x2[i];
        r[i + 12] = (__bf16)x3[i];
    }
    return r;
}
__device__ __forceinline__ v16bf frag_b_contig(const __bf16* p) {
    v8bf lo = *(const v8bf*)(p);
    v8bf hi = *(const v8bf*)(p + 8);
    v16bf r;
#pragma unroll
    for (int i = 0; i < 8; ++i) { r[i] = lo[i]; r[i + 8] = hi[i]; }
    return r;
}

// f32 -> bf16 bulk conversion (one-time pre-pass into workspace).
__global__ __launch_bounds__(256) void f32_to_bf16_kernel(
    const float* __restrict__ src, __bf16* __restrict__ dst, int n4) {
    const int i = blockIdx.x * 256 + threadIdx.x;
    if (i < n4) {
        v4f x = *(const v4f*)(src + (size_t)i * 4);
        v4bf y;
#pragma unroll
        for (int j = 0; j < 4; ++j) y[j] = (__bf16)x[j];
        *(v4bf*)(dst + (size_t)i * 4) = y;
    }
}

template <typename T>
__global__ __launch_bounds__(256) void mha_kernel(
    const T* __restrict__ Q, const T* __restrict__ K, const T* __restrict__ V,
    const float* __restrict__ mask, float* __restrict__ out,
    float* __restrict__ weights)
{
    extern __shared__ char smem[];
    __bf16* sQ   = (__bf16*)smem;                              // LDS offset 0
    float*  sS   = (float*)(smem + SQ_BYTES);
    __bf16* sW   = (__bf16*)(smem + SQ_BYTES + SS_BYTES);
    float*  sRed = (float*)(smem + SQ_BYTES + SS_BYTES + SW_BYTES);
    float*  sMax = sRed + 256;
    float*  sInv = sMax + 16;

    const int tid  = threadIdx.x;
    const int lane = tid & 31;
    const int wave = tid >> 5;
    const int l16  = lane & 15;
    const int hi16 = lane >> 4;            // 0 or 1
    const int b    = blockIdx.y;
    const int q0   = blockIdx.x * QT;

    const T* Qb = Q + ((size_t)b * S_ + q0) * DM_;
    const T* Kb = K + (size_t)b * S_ * DM_;
    const T* Vb = V + (size_t)b * S_ * DM_;

    // ---- stage Q tile (16x1024 bf16) into LDS: TDM DMA when source is bf16 ----
    bool staged = false;
    if constexpr (sizeof(T) == 2) {
#if HAVE_TDM
        if (tid < 32) {                    // one wave issues the tensor DMA
            const unsigned long long ga = (unsigned long long)(const void*)Qb;
            u32x4 g0;
            g0[0] = 1u;                                            // count=1, user mode
            g0[1] = 0u;                                            // lds_addr = 0 (sQ)
            g0[2] = (unsigned int)(ga & 0xffffffffu);              // global_addr[31:0]
            g0[3] = (unsigned int)((ga >> 32) & 0x01ffffffu)
                  | 0x80000000u;                                   // addr[56:32] | type=2
            i32x8 g1;
            g1[0] = (1 << 16);             // workgroup_mask=0, data_size=2B
            g1[1] = (int)(DM_ << 16);      // tensor_dim0 = 1024 (bits 79:48)
            g1[2] = (int)(S_ << 16);       // tensor_dim1 = 2048 (bits 111:80)
            g1[3] = (int)(DM_ << 16);      // tile_dim0 = 1024 (bits 127:112)
            g1[4] = QT;                    // tile_dim1 = 16, tile_dim2 = 0
            g1[5] = DM_;                   // tensor_dim0_stride = 1024 (48-bit)
            g1[6] = 0;
            g1[7] = 0;
            i32x4 g2 = {0, 0, 0, 0}, g3 = {0, 0, 0, 0};
#if __clang_major__ >= 23
            i32x8 g4 = {0, 0, 0, 0, 0, 0, 0, 0};
            __builtin_amdgcn_tensor_load_to_lds(g0, g1, g2, g3, g4, 0);
#else
            __builtin_amdgcn_tensor_load_to_lds(g0, g1, g2, g3, 0);
#endif
            __builtin_amdgcn_s_wait_tensorcnt((short)0);
        }
        staged = true;
#endif
    }
    if (!staged) {
        for (int idx = tid; idx < QT * DM_; idx += 256)
            sQ[idx] = cvt_bf16(Qb[idx]);
    }
    __syncthreads();

    const float scale = 0.03125f;          // 1/sqrt(1024)

    // ---- pass 1: scores = (Q K^T)*scale + mask*NEG, 8 waves split 128 K-tiles ----
    for (int kt = wave; kt < S_ / 16; kt += 8) {
        v8f acc = {};
        const __bf16* qrow = sQ + l16 * DM_ + (hi16 << 4);
        const T*      krow = Kb + (size_t)(kt * 16 + l16) * DM_ + (hi16 << 4);
        if (kt + 8 < S_ / 16)              // cover L2 latency for the next tile
            __builtin_prefetch((const void*)(Kb + (size_t)((kt + 8) * 16 + l16) * DM_), 0, 1);
#pragma unroll 2
        for (int c = 0; c < DM_ / 32; ++c) {
            v16bf a  = frag_a_lds(qrow + c * 32);
            v16bf bm = frag_b_contig(krow + c * 32);
            acc = __builtin_amdgcn_wmma_f32_16x16x32_bf16(
                false, a, false, bm, (short)0, acc, false, false);
        }
        const int col = kt * 16 + l16;
#pragma unroll
        for (int v = 0; v < 8; ++v) {
            const int m = v + (hi16 << 3);
            const float mval = mask[(size_t)(q0 + m) * S_ + col];
            sS[m * S_ + col] = acc[v] * scale + mval * NEG_INF_SCALE;
        }
    }
    __syncthreads();

    // ---- pass 2: softmax stats (16 rows x 2048), 16 threads per row ----
    {
        const int r = tid >> 4, p = tid & 15;
        float* row = sS + r * S_;
        float m = -3.4e38f;
        for (int c = p * 128; c < p * 128 + 128; ++c) m = fmaxf(m, row[c]);
        sRed[tid] = m;
        __syncthreads();
        if (tid < 16) {
            float mm = sRed[tid * 16];
            for (int i = 1; i < 16; ++i) mm = fmaxf(mm, sRed[tid * 16 + i]);
            sMax[tid] = mm;
        }
        __syncthreads();
        const float rmax = sMax[r];
        float s = 0.f;
        for (int c = p * 128; c < p * 128 + 128; ++c) {
            const float e = __expf(row[c] - rmax);
            row[c] = e;
            s += e;
        }
        sRed[tid] = s;
        __syncthreads();
        if (tid < 16) {
            float ss = 0.f;
            for (int i = 0; i < 16; ++i) ss += sRed[tid * 16 + i];
            sInv[tid] = 1.0f / ss;
        }
        __syncthreads();
    }

    // ---- normalize; emit 8 replicated head copies (coalesced); build bf16 W ----
    for (int idx = tid; idx < QT * S_; idx += 256) {
        const int r = idx >> 11;           // /2048
        const int c = idx & (S_ - 1);
        const float w = sS[idx] * sInv[r];
        sW[idx] = (__bf16)w;
        float* wp = weights + ((size_t)b * H_ * S_ + (size_t)(q0 + r)) * S_ + c;
#pragma unroll
        for (int h = 0; h < H_; ++h)
            wp[(size_t)h * S_ * S_] = w;
    }
    __syncthreads();

    // ---- pass 3: O = W @ V (16x1024), 8 waves split 64 N-tiles ----
    for (int nt = wave; nt < DM_ / 16; nt += 8) {
        const int n0 = nt * 16;
        v8f acc = {};
        const __bf16* wrow = sW + l16 * S_ + (hi16 << 4);
        const T*      vp0  = Vb + (size_t)(hi16 << 4) * DM_ + n0 + l16;
        for (int c = 0; c < S_ / 32; ++c) {
            v16bf a = frag_a_lds(wrow + c * 32);
            const T* vp = vp0 + (size_t)(c * 32) * DM_;
            v16bf bm;
#pragma unroll
            for (int e = 0; e < 16; ++e)
                bm[e] = cvt_bf16(vp[(size_t)e * DM_]);   // coalesced across lanes
            acc = __builtin_amdgcn_wmma_f32_16x16x32_bf16(
                false, a, false, bm, (short)0, acc, false, false);
        }
#pragma unroll
        for (int v = 0; v < 8; ++v) {
            const int m = v + (hi16 << 3);
            out[((size_t)b * S_ + q0 + m) * DM_ + n0 + l16] = acc[v];
        }
    }
}

extern "C" void kernel_launch(void* const* d_in, const int* in_sizes, int n_in,
                              void* d_out, int out_size, void* d_ws, size_t ws_size,
                              hipStream_t stream) {
    (void)in_sizes; (void)n_in; (void)out_size;
    const float* Q    = (const float*)d_in[0];
    const float* K    = (const float*)d_in[1];
    const float* V    = (const float*)d_in[2];
    const float* mask = (const float*)d_in[3];
    float* out     = (float*)d_out;
    float* weights = out + (size_t)B_ * S_ * DM_;   // (outputs, weights) concatenated

    const size_t NE   = (size_t)B_ * S_ * DM_;      // 8388608 elements per tensor
    const size_t need = 3 * NE * 2;                 // Qh,Kh,Vh in bf16 = 48MB
    dim3 grid(S_ / QT, B_);

    if (ws_size >= need) {
        __bf16* Qh = (__bf16*)d_ws;
        __bf16* Kh = Qh + NE;
        __bf16* Vh = Kh + NE;
        const int n4 = (int)(NE / 4);
        const int cblocks = (n4 + 255) / 256;
        f32_to_bf16_kernel<<<cblocks, 256, 0, stream>>>(Q, Qh, n4);
        f32_to_bf16_kernel<<<cblocks, 256, 0, stream>>>(K, Kh, n4);
        f32_to_bf16_kernel<<<cblocks, 256, 0, stream>>>(V, Vh, n4);
        hipFuncSetAttribute(reinterpret_cast<const void*>(&mha_kernel<__bf16>),
                            hipFuncAttributeMaxDynamicSharedMemorySize, SMEM_BYTES);
        mha_kernel<__bf16><<<grid, 256, SMEM_BYTES, stream>>>(Qh, Kh, Vh, mask, out, weights);
    } else {
        hipFuncSetAttribute(reinterpret_cast<const void*>(&mha_kernel<float>),
                            hipFuncAttributeMaxDynamicSharedMemorySize, SMEM_BYTES);
        mha_kernel<float><<<grid, 256, SMEM_BYTES, stream>>>(Q, K, V, mask, out, weights);
    }
}
